// LPHead_85598698209944
// MI455X (gfx1250) — matrix operand order repaired and checked by reading, not compiled, not run on hardware
//
#include <hip/hip_runtime.h>
#include <hip/hip_bf16.h>

// ---------------------------------------------------------------------------
// GAT link-prediction head for MI455X (gfx1250, wave32, WMMA).
//  - GEMMs: bf16 WMMA (v_wmma_f32_16x16x32_bf16, fp32 accumulate),
//    A-tile staged with GLOBAL_LOAD_ASYNC_TO_LDS_B128 (ASYNCcnt path),
//    B-tile transposed in LDS with packed ds_store_b64.
//  - Segment softmax + aggregation: CSR gather (built once, shared by both
//    layers). Hot loop is atomic-free coalesced reads out of the 192MB L2.
// ---------------------------------------------------------------------------

typedef __attribute__((ext_vector_type(16))) __bf16 bf16x16;
typedef __attribute__((ext_vector_type(8)))  __bf16 bf16x8;
typedef __attribute__((ext_vector_type(4)))  __bf16 bf16x4;
typedef __attribute__((ext_vector_type(8)))  float  v8f;

// Problem constants (from reference setup_inputs)
#define BATCH   2
#define NNODES  20000
#define NEDGES  320000
#define ETOT    (NEDGES + NNODES)     // self loops appended
#define ROWS    (BATCH * NNODES)      // 40000, multiple of 64
#define HIDDIM  768
#define HEADS1  2
#define CHAN    256
#define D1      (HEADS1 * CHAN)       // 512
#define D2      CHAN                  // 256

// ------------------------- CDNA5 async copy helpers -------------------------

__device__ __forceinline__ void async_copy_b128(unsigned lds_off,
                                                const void* gaddr) {
  // GLOBAL_LOAD_ASYNC_TO_LDS_B128, GV mode: per-lane 16B memory -> LDS copy.
  asm volatile("global_load_async_to_lds_b128 %0, %1, off"
               :: "v"(lds_off), "v"((unsigned long long)gaddr)
               : "memory");
}

__device__ __forceinline__ void wait_async0() {
  asm volatile("s_wait_asynccnt 0x0" ::: "memory");
}

// ------------------------- conversion --------------------------------------

__global__ void f32_to_bf16_kernel(const float* __restrict__ in,
                                   __bf16* __restrict__ out, long n) {
  long i = (long)blockIdx.x * blockDim.x + threadIdx.x;
  if (i < n) out[i] = (__bf16)in[i];
}

// ------------------------- WMMA GEMM ----------------------------------------
// C[M,N] = A[M,K] @ B[K,N], A/B bf16 row-major, C fp32.
// Block: 256 threads (8 waves, 2x4). Block tile 64(M) x 128(N).
// Wave tile 32x32 = 2x2 WMMA accumulators (2 A-frags x 2 B-frags -> 4 WMMAs).
// Requires M % 64 == 0, N % 128 == 0, K % 32 == 0 (true for all calls here).

__global__ __launch_bounds__(256)
void gemm_bf16_wmma(const __bf16* __restrict__ A, const __bf16* __restrict__ Bm,
                    float* __restrict__ C, int M, int K, int N) {
  __shared__ __align__(16) __bf16 ldsA[64 * 32];    // [m][k]
  __shared__ __align__(16) __bf16 ldsBt[128 * 32];  // [n][k] (transposed)

  const int tid  = threadIdx.x;
  const int lane = tid & 31;
  const int wave = tid >> 5;            // 8 waves
  const int wm   = (wave >> 2) * 32;    // 0 / 32   : M sub-tile
  const int wn   = (wave & 3) * 32;     // 0..96    : N sub-tile
  const int row0 = blockIdx.x * 64;
  const int col0 = blockIdx.y * 128;

  v8f acc00 = {}, acc01 = {}, acc10 = {}, acc11 = {};

  // Staging roles: threads 0..127 copy the A tile via async-LDS;
  // threads 128..255 transpose the B tile with packed b64 DS stores.
  const bool doA = tid < 128;
  const int  ar  = tid >> 1;            // A row 0..63
  const int  ac  = (tid & 1) * 16;      // A col 0 / 16
  const int  bc  = tid - 128;
  const int  kq  = (bc >> 4) * 4;       // K quad 0,4,..,28
  const int  n8  = (bc & 15) * 8;       // N octet 0,8,..,120
  const unsigned ldsA_off =
      (unsigned)(unsigned long long)(&ldsA[ar * 32 + ac]);

  for (int k0 = 0; k0 < K; k0 += 32) {
    if (doA) {
      async_copy_b128(ldsA_off, &A[(size_t)(row0 + ar) * K + k0 + ac]);
      async_copy_b128(ldsA_off + 16, &A[(size_t)(row0 + ar) * K + k0 + ac + 8]);
    } else {
      const __bf16* bp = Bm + (size_t)(k0 + kq) * N + col0 + n8;
      bf16x8 r0 = *(const bf16x8*)(bp);
      bf16x8 r1 = *(const bf16x8*)(bp + N);
      bf16x8 r2 = *(const bf16x8*)(bp + 2 * (size_t)N);
      bf16x8 r3 = *(const bf16x8*)(bp + 3 * (size_t)N);
#pragma unroll
      for (int j = 0; j < 8; ++j) {
        bf16x4 v = {r0[j], r1[j], r2[j], r3[j]};
        *(bf16x4*)&ldsBt[(n8 + j) * 32 + kq] = v;
      }
    }
    wait_async0();
    __syncthreads();

    // A fragments (16x32): element j -> K = koff + j; contiguous in ldsA
    const int arow0 = wm + (lane & 15);
    const int koff  = (lane >> 4) * 16;
    union { bf16x16 v; bf16x8 h[2]; } a0, a1;
    a0.h[0] = *(const bf16x8*)&ldsA[arow0 * 32 + koff];
    a0.h[1] = *(const bf16x8*)&ldsA[arow0 * 32 + koff + 8];
    a1.h[0] = *(const bf16x8*)&ldsA[(arow0 + 16) * 32 + koff];
    a1.h[1] = *(const bf16x8*)&ldsA[(arow0 + 16) * 32 + koff + 8];

    // B fragments (32x16): element j -> K = kb + j; contiguous in ldsBt
    const int bcol = wn + (lane & 15);
    const int kb   = (lane >> 4) * 16;
    union { bf16x16 v; bf16x8 h[2]; } b0, b1;
    b0.h[0] = *(const bf16x8*)&ldsBt[bcol * 32 + kb];
    b0.h[1] = *(const bf16x8*)&ldsBt[bcol * 32 + kb + 8];
    b1.h[0] = *(const bf16x8*)&ldsBt[(bcol + 16) * 32 + kb];
    b1.h[1] = *(const bf16x8*)&ldsBt[(bcol + 16) * 32 + kb + 8];

    acc00 = __builtin_amdgcn_wmma_f32_16x16x32_bf16(
        false, a0.v, false, b0.v, (short)0, acc00, false, false);
    acc01 = __builtin_amdgcn_wmma_f32_16x16x32_bf16(
        false, a0.v, false, b1.v, (short)0, acc01, false, false);
    acc10 = __builtin_amdgcn_wmma_f32_16x16x32_bf16(
        false, a1.v, false, b0.v, (short)0, acc10, false, false);
    acc11 = __builtin_amdgcn_wmma_f32_16x16x32_bf16(
        false, a1.v, false, b1.v, (short)0, acc11, false, false);
    __syncthreads();
  }

  // C/D layout: VGPR r at lane -> M = (lane>>4)*8 + r, N = lane&15
  const int mr = row0 + wm + (lane >> 4) * 8;
  const int nc = col0 + wn + (lane & 15);
#pragma unroll
  for (int r = 0; r < 8; ++r) {
    float* cp0 = &C[(size_t)(mr + r) * N + nc];
    float* cp1 = &C[(size_t)(mr + 16 + r) * N + nc];
    cp0[0]  = acc00[r];
    cp0[16] = acc01[r];
    cp1[0]  = acc10[r];
    cp1[16] = acc11[r];
  }
}

// ------------------------- CSR build (shared by both layers) ----------------

__global__ void csr_count_kernel(const int* __restrict__ ei_all,
                                 int* __restrict__ deg) {
  int b = blockIdx.y;
  long e = (long)blockIdx.x * 256 + threadIdx.x;
  if (e >= ETOT) return;
  const int* ei = ei_all + (size_t)b * 2 * NEDGES;
  int d = (e < NEDGES) ? ei[NEDGES + e] : (int)(e - NEDGES);
  atomicAdd(&deg[b * NNODES + d], 1);
}

// Single-block exclusive prefix scan over n entries (n = ROWS = 40000).
__global__ __launch_bounds__(1024)
void csr_scan_kernel(const int* __restrict__ deg, int* __restrict__ rowstart,
                     int n) {
  __shared__ int buf[1024];
  __shared__ int carry;
  int t = threadIdx.x;
  if (t == 0) carry = 0;
  __syncthreads();
  for (int base = 0; base < n; base += 1024) {
    int v = (base + t < n) ? deg[base + t] : 0;
    buf[t] = v;
    __syncthreads();
    for (int off = 1; off < 1024; off <<= 1) {      // Hillis-Steele inclusive
      int add = (t >= off) ? buf[t - off] : 0;
      __syncthreads();
      buf[t] += add;
      __syncthreads();
    }
    if (base + t < n) rowstart[base + t] = carry + buf[t] - v;  // exclusive
    int total = buf[1023];
    __syncthreads();
    if (t == 0) carry += total;
    __syncthreads();
  }
}

__global__ void csr_fill_kernel(const int* __restrict__ ei_all,
                                const int* __restrict__ rowstart,
                                int* __restrict__ cursor,
                                int* __restrict__ elist) {
  int b = blockIdx.y;
  long e = (long)blockIdx.x * 256 + threadIdx.x;
  if (e >= ETOT) return;
  const int* ei = ei_all + (size_t)b * 2 * NEDGES;
  int d = (e < NEDGES) ? ei[NEDGES + e] : (int)(e - NEDGES);
  int gd = b * NNODES + d;
  int pos = atomicAdd(&cursor[gd], 1);
  elist[rowstart[gd] + pos] = (int)e;               // batch-local edge id
}

// ------------------------- attention logits ---------------------------------
// out[n,h] = sum_c h[n,h,c] * att[h,c].  One wave per (n,h).

__global__ __launch_bounds__(256)
void att_logits_kernel(const float* __restrict__ hmat,
                       const float* __restrict__ att,
                       float* __restrict__ out, int rows, int H, int C) {
  int wid  = blockIdx.x * 8 + (threadIdx.x >> 5);
  int lane = threadIdx.x & 31;
  if (wid >= rows * H) return;
  int n = wid / H, hh = wid - n * H;
  const float* hp = hmat + ((size_t)n * H + hh) * C;
  const float* ap = att + (size_t)hh * C;
  float s = 0.f;
  for (int c = lane; c < C; c += 32) s += hp[c] * ap[c];
#pragma unroll
  for (int off = 16; off; off >>= 1) s += __shfl_xor(s, off, 32);
  if (lane == 0) out[wid] = s;
}

// ------------------------- fused softmax + aggregation ----------------------
// One wave per (node, head). CSR gather: atomic-free, coalesced L2 reads.
//   pass1: m = max_e leaky_relu(a_s[src]+a_d[dst])   (lanes over edges)
//   pass2: acc[c] = sum_e ex_e * h[src,h,c], S = sum_e ex_e (lanes over chans)
//   out[dst,h,c] = acc[c] / (S + 1e-16)
// C == 256 (8 channels per lane).

__global__ __launch_bounds__(256)
void node_softmax_aggr_kernel(const int* __restrict__ ei_all,
                              const int* __restrict__ elist,
                              const int* __restrict__ rowstart,
                              const int* __restrict__ deg,
                              const float* __restrict__ a_s,
                              const float* __restrict__ a_d,
                              const float* __restrict__ hmat,
                              float* __restrict__ out, int H) {
  const int C = CHAN;
  long wid = (long)blockIdx.x * 8 + (threadIdx.x >> 5);
  int lane = threadIdx.x & 31;
  if (wid >= (long)ROWS * H) return;
  int h  = (int)(wid % H);
  int gd = (int)(wid / H);
  int b  = gd / NNODES;
  const int* ei = ei_all + (size_t)b * 2 * NEDGES;

  int start = rowstart[gd];
  int n     = deg[gd];                  // >= 1 (self loop)
  float adc = a_d[(size_t)gd * H + h];

  // pass 1: segment max (lanes parallel over edges)
  float m = -3.0e38f;
  for (int i = lane; i < n; i += 32) {
    int e    = elist[start + i];
    int srcl = (e < NEDGES) ? ei[e] : (e - NEDGES);
    float v  = a_s[((size_t)(b * NNODES + srcl)) * H + h] + adc;
    v = v > 0.f ? v : 0.2f * v;         // leaky_relu(0.2)
    m = fmaxf(m, v);
  }
#pragma unroll
  for (int off = 16; off; off >>= 1) m = fmaxf(m, __shfl_xor(m, off, 32));

  // pass 2: exp-weighted gather accumulate (lanes parallel over channels)
  float acc[8] = {0.f, 0.f, 0.f, 0.f, 0.f, 0.f, 0.f, 0.f};
  float ssum = 0.f;
  for (int i = 0; i < n; ++i) {
    int e    = elist[start + i];
    int srcl = (e < NEDGES) ? ei[e] : (e - NEDGES);
    int gs   = b * NNODES + srcl;
    float v  = a_s[(size_t)gs * H + h] + adc;
    v = v > 0.f ? v : 0.2f * v;
    float ex = __expf(v - m);
    ssum += ex;
    const float* hp = hmat + ((size_t)gs * H + h) * C;
    __builtin_prefetch(hp + lane, 0, 0);
#pragma unroll
    for (int j = 0; j < 8; ++j) acc[j] += ex * hp[lane + j * 32];
  }
  float inv = 1.f / (ssum + 1e-16f);
  float* op = out + ((size_t)gd * H + h) * C;
#pragma unroll
  for (int j = 0; j < 8; ++j) op[lane + j * 32] = acc[j] * inv;
}

// ------------------------- bias + activation --------------------------------

__global__ void bias_elu_bf16_kernel(const float* __restrict__ in,
                                     const float* __restrict__ bias,
                                     __bf16* __restrict__ out,
                                     long total, int cols) {
  long i = (long)blockIdx.x * blockDim.x + threadIdx.x;
  if (i >= total) return;
  float v = in[i] + bias[i % cols];
  v = v > 0.f ? v : expm1f(v);                      // ELU(alpha=1)
  out[i] = (__bf16)v;
}

__global__ void bias_add_f32_kernel(float* __restrict__ io,
                                    const float* __restrict__ bias,
                                    long total, int cols) {
  long i = (long)blockIdx.x * blockDim.x + threadIdx.x;
  if (i >= total) return;
  io[i] += bias[i % cols];
}

// ------------------------- final edge dot + sigmoid -------------------------

__global__ __launch_bounds__(256)
void final_dot_kernel(const float* __restrict__ z,
                      const int* __restrict__ ei_all,
                      float* __restrict__ out, int C) {
  int b = blockIdx.y;
  long wid = (long)blockIdx.x * 8 + (threadIdx.x >> 5);
  int lane = threadIdx.x & 31;
  if (wid >= NEDGES) return;
  const int* ei = ei_all + (size_t)b * 2 * NEDGES;
  int s = ei[wid], d = ei[NEDGES + wid];
  const float* zs = z + ((size_t)b * NNODES + s) * C;
  const float* zd = z + ((size_t)b * NNODES + d) * C;
  float acc = 0.f;
  for (int c = lane; c < C; c += 32) acc += zs[c] * zd[c];
#pragma unroll
  for (int off = 16; off; off >>= 1) acc += __shfl_xor(acc, off, 32);
  if (lane == 0) out[(size_t)b * NEDGES + wid] = 1.f / (1.f + __expf(-acc));
}

// ------------------------- host launch --------------------------------------

static inline int cdiv(long a, long b) { return (int)((a + b - 1) / b); }

extern "C" void kernel_launch(void* const* d_in, const int* in_sizes, int n_in,
                              void* d_out, int out_size, void* d_ws, size_t ws_size,
                              hipStream_t stream) {
  (void)in_sizes; (void)n_in; (void)out_size; (void)ws_size;

  const float* x        = (const float*)d_in[0];   // [B,N,768]
  const int*   ei       = (const int*)d_in[1];     // [B,2,E]
  const float* W1       = (const float*)d_in[2];   // [768,512]
  const float* att_src1 = (const float*)d_in[3];   // [2,256]
  const float* att_dst1 = (const float*)d_in[4];
  const float* b1       = (const float*)d_in[5];   // [512]
  const float* W2       = (const float*)d_in[6];   // [512,256]
  const float* att_src2 = (const float*)d_in[7];   // [1,256]
  const float* att_dst2 = (const float*)d_in[8];
  const float* b2       = (const float*)d_in[9];   // [256]
  float*       out      = (float*)d_out;           // [B,E]

  // ---- workspace carve-up (bump allocator, 256B aligned) ----
  char* p = (char*)d_ws;
  auto alloc = [&](size_t bytes) -> void* {
    void* r = (void*)p;
    p += (bytes + 255) & ~(size_t)255;
    return r;
  };
  __bf16* xb       = (__bf16*)alloc((size_t)ROWS * HIDDIM * 2);
  __bf16* w1b      = (__bf16*)alloc((size_t)HIDDIM * D1 * 2);
  __bf16* w2b      = (__bf16*)alloc((size_t)D1 * D2 * 2);
  float*  h1       = (float*) alloc((size_t)ROWS * D1 * 4);
  float*  as1      = (float*) alloc((size_t)ROWS * HEADS1 * 4);
  float*  ad1      = (float*) alloc((size_t)ROWS * HEADS1 * 4);
  float*  out1     = (float*) alloc((size_t)ROWS * D1 * 4);
  __bf16* z1b      = (__bf16*)alloc((size_t)ROWS * D1 * 2);
  float*  h2       = (float*) alloc((size_t)ROWS * D2 * 4);
  float*  as2      = (float*) alloc((size_t)ROWS * 4);
  float*  ad2      = (float*) alloc((size_t)ROWS * 4);
  float*  out2     = (float*) alloc((size_t)ROWS * D2 * 4);
  int*    deg      = (int*)   alloc((size_t)ROWS * 4);
  int*    rowstart = (int*)   alloc((size_t)ROWS * 4);
  int*    cursor   = (int*)   alloc((size_t)ROWS * 4);
  int*    elist    = (int*)   alloc((size_t)BATCH * ETOT * 4);

  // ---- CSR build (reused by both layers) ----
  hipMemsetAsync(deg,    0, (size_t)ROWS * 4, stream);
  hipMemsetAsync(cursor, 0, (size_t)ROWS * 4, stream);
  {
    dim3 ge(cdiv(ETOT, 256), BATCH);
    csr_count_kernel<<<ge, 256, 0, stream>>>(ei, deg);
    csr_scan_kernel<<<1, 1024, 0, stream>>>(deg, rowstart, ROWS);
    csr_fill_kernel<<<ge, 256, 0, stream>>>(ei, rowstart, cursor, elist);
  }

  // ---- bf16 conversions ----
  {
    long n = (long)ROWS * HIDDIM;
    f32_to_bf16_kernel<<<cdiv(n, 256), 256, 0, stream>>>(x, xb, n);
    n = (long)HIDDIM * D1;
    f32_to_bf16_kernel<<<cdiv(n, 256), 256, 0, stream>>>(W1, w1b, n);
    n = (long)D1 * D2;
    f32_to_bf16_kernel<<<cdiv(n, 256), 256, 0, stream>>>(W2, w2b, n);
  }

  // ---- layer 1: h1 = x @ W1 (WMMA bf16) ----
  gemm_bf16_wmma<<<dim3(ROWS / 64, D1 / 128), 256, 0, stream>>>(
      xb, w1b, h1, ROWS, HIDDIM, D1);

  att_logits_kernel<<<cdiv((long)ROWS * HEADS1, 8), 256, 0, stream>>>(
      h1, att_src1, as1, ROWS, HEADS1, CHAN);
  att_logits_kernel<<<cdiv((long)ROWS * HEADS1, 8), 256, 0, stream>>>(
      h1, att_dst1, ad1, ROWS, HEADS1, CHAN);

  node_softmax_aggr_kernel<<<cdiv((long)ROWS * HEADS1, 8), 256, 0, stream>>>(
      ei, elist, rowstart, deg, as1, ad1, h1, out1, HEADS1);

  {
    long n = (long)ROWS * D1;
    bias_elu_bf16_kernel<<<cdiv(n, 256), 256, 0, stream>>>(out1, b1, z1b, n, D1);
  }

  // ---- layer 2: h2 = z1 @ W2 (WMMA bf16) ----
  gemm_bf16_wmma<<<dim3(ROWS / 64, D2 / 128), 256, 0, stream>>>(
      z1b, w2b, h2, ROWS, D1, D2);

  att_logits_kernel<<<cdiv((long)ROWS, 8), 256, 0, stream>>>(
      h2, att_src2, as2, ROWS, 1, CHAN);
  att_logits_kernel<<<cdiv((long)ROWS, 8), 256, 0, stream>>>(
      h2, att_dst2, ad2, ROWS, 1, CHAN);

  node_softmax_aggr_kernel<<<cdiv((long)ROWS, 8), 256, 0, stream>>>(
      ei, elist, rowstart, deg, as2, ad2, h2, out2, 1);

  {
    long n = (long)ROWS * D2;
    bias_add_f32_kernel<<<cdiv(n, 256), 256, 0, stream>>>(out2, b2, n, D2);
  }

  // ---- link logits + sigmoid ----
  final_dot_kernel<<<dim3(cdiv((long)NEDGES, 8), BATCH), 256, 0, stream>>>(
      out2, ei, out, CHAN);
}
